// LegacyRelPositionMultiHeadedAttention_73710228734781
// MI455X (gfx1250) — matrix operand order repaired and checked by reading, not compiled, not run on hardware
//
#include <hip/hip_runtime.h>

// ---------------------------------------------------------------------------
// LegacyRelPositionMultiHeadedAttention for MI455X (gfx1250, wave32, WMMA).
// B=8, T=1024, D=512, H=8, DK=64.
// All GEMMs + attention einsums via v_wmma_f32_16x16x32_bf16 (f32 accum).
// Flash-style streaming softmax: the 256MB (B,H,T,T) score tensor is never
// materialized (would add ~30us of HBM traffic at 23.3 TB/s vs ~3us for I/O).
// Workspace (bf16): q_u, q_v, K, V (8.4MB ea), P (1MB), x (8.4MB) ~= 43MB.
//
// Round-3 change: softmax row reductions moved from ds_swizzle (DS-unit op +
// s_wait_dscnt 0 per step, fully serialized) to DPP16 row_ror butterflies —
// pure VALU, no counters, co-issues with the WMMA (XDL/TRANS) pipeline.
// ---------------------------------------------------------------------------

#define B_ 8
#define T_ 1024
#define D_ 512
#define H_ 8
#define DK_ 64

typedef __attribute__((ext_vector_type(4)))  float  v4f;
typedef __attribute__((ext_vector_type(8)))  float  v8f;
typedef __attribute__((ext_vector_type(4)))  __bf16 v4bf;
typedef __attribute__((ext_vector_type(8)))  __bf16 v8bf;
typedef __attribute__((ext_vector_type(16))) __bf16 v16bf;

__device__ __forceinline__ v8f wmma_bf16(v16bf a, v16bf b, v8f c) {
  // (neg_a, A, neg_b, B, c_mod, C, reuse_a, reuse_b)
  return __builtin_amdgcn_wmma_f32_16x16x32_bf16(false, a, false, b, (short)0,
                                                 c, false, false);
}

__device__ __forceinline__ v8f vzero8() {
  v8f z;
#pragma unroll
  for (int i = 0; i < 8; ++i) z[i] = 0.0f;
  return z;
}

// Assemble a 16-element bf16 fragment from two 8-halves (16B) chunks.
__device__ __forceinline__ v16bf ld_frag(const __bf16* lo8, const __bf16* hi8) {
  v8bf lo = *(const v8bf*)lo8;
  v8bf hi = *(const v8bf*)hi8;
  v16bf r;
#pragma unroll
  for (int i = 0; i < 8; ++i) { r[i] = lo[i]; r[i + 8] = hi[i]; }
  return r;
}

__device__ __forceinline__ v4bf cvt4(v4f x) {
  v4bf r;
#pragma unroll
  for (int i = 0; i < 4; ++i) r[i] = (__bf16)x[i];  // -> v_cvt_pk_bf16_f32
  return r;
}

// 16-lane rotation butterflies via DPP16 (row_ror:N = dpp_ctrl 0x120|N).
// Pure VALU (v_mov_b32_dpp) — no DS unit, no counter waits.
template <int CTRL>
__device__ __forceinline__ float dpp_rot(float v) {
  return __int_as_float(__builtin_amdgcn_update_dpp(
      0, __float_as_int(v), CTRL, 0xf, 0xf, true));
}
__device__ __forceinline__ float red_max16(float v) {
  v = fmaxf(v, dpp_rot<0x128>(v));  // row_ror:8
  v = fmaxf(v, dpp_rot<0x124>(v));  // row_ror:4
  v = fmaxf(v, dpp_rot<0x122>(v));  // row_ror:2
  v = fmaxf(v, dpp_rot<0x121>(v));  // row_ror:1
  return v;
}
__device__ __forceinline__ float red_sum16(float v) {
  v += dpp_rot<0x128>(v);
  v += dpp_rot<0x124>(v);
  v += dpp_rot<0x122>(v);
  v += dpp_rot<0x121>(v);
  return v;
}

// ---------------------------------------------------------------------------
// Tiled GEMM:  Y[m][n] = sum_k X[m][k] * W[n][k] + bias[n]
// Block tile 128x64, 8 waves; each wave: 16 rows x 64 cols (4 accumulators).
// K staged to LDS in 32-steps, f32->bf16 conversion during staging.
// XMODE: 0 = X is f32, 1 = X is bf16.
// OMODE bitmask: 1 = outA(+biasU) bf16, 2 = outB(+biasV) bf16, 4 = outF f32.
// ---------------------------------------------------------------------------
template <int XMODE, int OMODE>
__global__ __launch_bounds__(256) void proj_gemm_k(
    const float* __restrict__ Xf, const __bf16* __restrict__ Xb,
    const float* __restrict__ W, const float* __restrict__ bias,
    const float* __restrict__ biasU, const float* __restrict__ biasV,
    __bf16* __restrict__ outA, __bf16* __restrict__ outB,
    float* __restrict__ outF) {
  constexpr int TM = 128, TN = 64, KC = 32, K = D_;
  __shared__ __bf16 la[TM][KC];  // 8 KB
  __shared__ __bf16 lb[TN][KC];  // 4 KB

  const int tid = threadIdx.x;
  const int wave = tid >> 5, lane = tid & 31;
  const int lh = lane >> 4, l16 = lane & 15;
  const int mt = blockIdx.y * TM, nt = blockIdx.x * TN;

  v8f acc[4];
#pragma unroll
  for (int i = 0; i < 4; ++i) acc[i] = vzero8();

  for (int kb = 0; kb < K; kb += KC) {
    // ---- Stage A tile (128 x 32) ----
    if (XMODE == 0) {
      // 4096 f32 = 1024 float4 groups; 4 groups per thread.
#pragma unroll
      for (int i = 0; i < 4; ++i) {
        int idx = i * 256 + tid;
        int r = idx >> 3, c = (idx & 7) * 4;  // 8 float4 per row
        v4f x = *(const v4f*)(Xf + (size_t)(mt + r) * K + kb + c);
        *(v4bf*)&la[r][c] = cvt4(x);
      }
    } else {
      // 4096 bf16 = 512 v8bf groups; 2 groups per thread.
#pragma unroll
      for (int i = 0; i < 2; ++i) {
        int idx = i * 256 + tid;
        int r = idx >> 2, c = (idx & 3) * 8;  // 4 v8bf per row
        *(v8bf*)&la[r][c] = *(const v8bf*)(Xb + (size_t)(mt + r) * K + kb + c);
      }
    }
    // ---- Stage B tile: W[n][k] (64 x 32), 2048 f32 = 512 float4 groups ----
#pragma unroll
    for (int i = 0; i < 2; ++i) {
      int idx = i * 256 + tid;
      int n = idx >> 3, c = (idx & 7) * 4;
      v4f w = *(const v4f*)(W + (size_t)(nt + n) * K + kb + c);
      *(v4bf*)&lb[n][c] = cvt4(w);
    }
    __syncthreads();

    // A-layout (16x32): lane -> M=l16; K = lh*16 + e (16 contiguous).
    v16bf af = ld_frag(&la[wave * 16 + l16][lh * 16],
                       &la[wave * 16 + l16][lh * 16 + 8]);
#pragma unroll
    for (int ns = 0; ns < 4; ++ns) {
      // B-layout (32x16): lane -> N=l16; K = lh*16 + e (16 contiguous).
      v16bf bf = ld_frag(&lb[ns * 16 + l16][lh * 16],
                         &lb[ns * 16 + l16][lh * 16 + 8]);
      acc[ns] = wmma_bf16(af, bf, acc[ns]);
    }
    __syncthreads();
  }

  // Epilogue. C-layout: lane holds C[m = j + 8*lh][n = l16] in VGPR j.
#pragma unroll
  for (int ns = 0; ns < 4; ++ns) {
    int n = nt + ns * 16 + l16;
    float b0 = bias ? bias[n] : 0.0f;
    float bu = (OMODE & 1) && biasU ? biasU[n] : 0.0f;
    float bv = (OMODE & 2) && biasV ? biasV[n] : 0.0f;
#pragma unroll
    for (int j = 0; j < 8; ++j) {
      size_t m = (size_t)mt + wave * 16 + j + 8 * lh;
      float v = acc[ns][j] + b0;
      if (OMODE & 1) outA[m * K + n] = (__bf16)(v + bu);
      if (OMODE & 2) outB[m * K + n] = (__bf16)(v + bv);
      if (OMODE & 4) outF[m * K + n] = v;
    }
  }
}

// ---------------------------------------------------------------------------
// Streaming (flash-style) relative-position attention.
// Grid: (T/64 row-tiles, B*H). Block: 128 threads = 4 waves, 16 q-rows/wave.
// Per 32-key block: S = q_u.K^T + q_v.P^T (8 WMMAs), scale+mask, online
// softmax (DPP row reductions), probs -> per-wave LDS (C-layout -> A-layout
// reshape), O += P.V (4 WMMAs against LDS-transposed V).
// ---------------------------------------------------------------------------
__global__ __launch_bounds__(128) void attn_k(
    const __bf16* __restrict__ qu, const __bf16* __restrict__ qv,
    const __bf16* __restrict__ kmat, const __bf16* __restrict__ pmat,
    const __bf16* __restrict__ vmat, const int* __restrict__ mask,
    __bf16* __restrict__ xout) {
  constexpr int KB = 32;
  __shared__ __bf16 lk[KB][DK_];       // K block   (4 KB)
  __shared__ __bf16 lp[KB][DK_];       // P block   (4 KB)
  __shared__ __bf16 lvt[DK_][KB];      // V block, transposed (4 KB)
  __shared__ float lvalid[KB];
  __shared__ __bf16 lprob[4][16][KB];  // per-wave prob staging (4 KB)

  const int tid = threadIdx.x;
  const int wave = tid >> 5, lane = tid & 31;
  const int lh = lane >> 4, l16 = lane & 15;
  const int b = blockIdx.y >> 3, h = blockIdx.y & 7;
  const int q0 = blockIdx.x * 64 + wave * 16;

  // Q fragments (A-layout) straight from global bf16 workspace.
  const __bf16* qbu = qu + ((size_t)(b * T_ + q0 + l16)) * D_ + h * DK_;
  const __bf16* qbv = qv + ((size_t)(b * T_ + q0 + l16)) * D_ + h * DK_;
  v16bf fu[2], fv[2];
#pragma unroll
  for (int c = 0; c < 2; ++c) {
    int k0 = c * 32 + lh * 16;
    fu[c] = ld_frag(qbu + k0, qbu + k0 + 8);
    fv[c] = ld_frag(qbv + k0, qbv + k0 + 8);
  }

  v8f oacc[4];
#pragma unroll
  for (int c = 0; c < 4; ++c) oacc[c] = vzero8();
  float mst[8], lst[8];
#pragma unroll
  for (int j = 0; j < 8; ++j) { mst[j] = -3.0e38f; lst[j] = 0.0f; }

  for (int kb = 0; kb < T_; kb += KB) {
    // Stage K / P / V(transposed), 16B-vectorized: 32x64 halves = 256 v8bf
    // groups; 2 groups per thread per matrix.
#pragma unroll
    for (int i = 0; i < 2; ++i) {
      int idx = i * 128 + tid;
      int ky = idx >> 3, d8 = (idx & 7) * 8;
      size_t gk = ((size_t)(b * T_ + kb + ky)) * D_ + h * DK_ + d8;
      v8bf kv = *(const v8bf*)(kmat + gk);
      v8bf pv = *(const v8bf*)(pmat + ((size_t)(kb + ky)) * D_ + h * DK_ + d8);
      v8bf vv = *(const v8bf*)(vmat + gk);
      *(v8bf*)&lk[ky][d8] = kv;
      *(v8bf*)&lp[ky][d8] = pv;
#pragma unroll
      for (int e = 0; e < 8; ++e) lvt[d8 + e][ky] = vv[e];  // transpose
    }
    if (tid < KB) lvalid[tid] = mask[b * T_ + kb + tid] ? 1.0f : 0.0f;
    if (kb + KB < T_)  // hint next tile toward L2 (global_prefetch_b8)
      __builtin_prefetch(
          kmat + ((size_t)(b * T_ + kb + KB)) * D_ + h * DK_ + tid, 0, 1);
    __syncthreads();

    // Scores: two 16-key sub-tiles, DK=64 -> two K=32 WMMA chunks each.
    v8f s[2];
    float vf[2];
#pragma unroll
    for (int ks = 0; ks < 2; ++ks) {
      v8f sa = vzero8();
#pragma unroll
      for (int c = 0; c < 2; ++c) {
        v16bf bk = ld_frag(&lk[ks * 16 + l16][c * 32 + lh * 16],
                           &lk[ks * 16 + l16][c * 32 + lh * 16 + 8]);
        sa = wmma_bf16(fu[c], bk, sa);
      }
#pragma unroll
      for (int c = 0; c < 2; ++c) {
        v16bf bp = ld_frag(&lp[ks * 16 + l16][c * 32 + lh * 16],
                           &lp[ks * 16 + l16][c * 32 + lh * 16 + 8]);
        sa = wmma_bf16(fv[c], bp, sa);
      }
      s[ks] = sa;
      vf[ks] = lvalid[ks * 16 + l16];
    }

    // Online softmax update (per row j; cols live across 16 lanes).
    float p0[8], p1[8], alpha[8];
#pragma unroll
    for (int j = 0; j < 8; ++j) {
      float s0 = s[0][j] * 0.125f;  // 1/sqrt(DK)
      float s1 = s[1][j] * 0.125f;
      float t0 = vf[0] > 0.5f ? s0 : -3.0e38f;
      float t1 = vf[1] > 0.5f ? s1 : -3.0e38f;
      float bm = red_max16(fmaxf(t0, t1));
      float mn = fmaxf(mst[j], bm);
      float al = __expf(mst[j] - mn);
      float e0 = vf[0] > 0.5f ? __expf(s0 - mn) : 0.0f;  // exact post-mask zero
      float e1 = vf[1] > 0.5f ? __expf(s1 - mn) : 0.0f;
      float rs = red_sum16(e0 + e1);
      lst[j] = lst[j] * al + rs;
      mst[j] = mn;
      alpha[j] = al;
      p0[j] = e0;
      p1[j] = e1;
    }
#pragma unroll
    for (int c = 0; c < 4; ++c)
#pragma unroll
      for (int j = 0; j < 8; ++j) oacc[c][j] *= alpha[j];

    // Reshape probs C-layout -> A-layout through per-wave LDS.
#pragma unroll
    for (int j = 0; j < 8; ++j) {
      lprob[wave][j + 8 * lh][l16] = (__bf16)p0[j];
      lprob[wave][j + 8 * lh][16 + l16] = (__bf16)p1[j];
    }
    asm volatile("s_wait_dscnt 0" ::: "memory");  // wave-local LDS visibility

    v16bf fp = ld_frag(&lprob[wave][l16][lh * 16],
                       &lprob[wave][l16][lh * 16 + 8]);
#pragma unroll
    for (int c = 0; c < 4; ++c) {
      v16bf bv = ld_frag(&lvt[c * 16 + l16][lh * 16],
                         &lvt[c * 16 + l16][lh * 16 + 8]);
      oacc[c] = wmma_bf16(fp, bv, oacc[c]);
    }
    __syncthreads();
  }

  // Epilogue: x[b, q, h*64+dk] = O / l   (l==0 -> 0, matches re-masking).
#pragma unroll
  for (int j = 0; j < 8; ++j) {
    float inv = lst[j] > 0.0f ? 1.0f / lst[j] : 0.0f;
    size_t row = (size_t)b * T_ + q0 + j + 8 * lh;
#pragma unroll
    for (int c = 0; c < 4; ++c)
      xout[row * D_ + h * DK_ + c * 16 + l16] = (__bf16)(oacc[c][j] * inv);
  }
}

// ---------------------------------------------------------------------------
extern "C" void kernel_launch(void* const* d_in, const int* in_sizes, int n_in,
                              void* d_out, int out_size, void* d_ws,
                              size_t ws_size, hipStream_t stream) {
  (void)in_sizes; (void)n_in; (void)out_size; (void)ws_size;
  const float* query = (const float*)d_in[0];
  const float* key_  = (const float*)d_in[1];
  const float* value = (const float*)d_in[2];
  const int*   maskp = (const int*)d_in[3];
  const float* pos   = (const float*)d_in[4];
  const float* Wq = (const float*)d_in[5];
  const float* bq = (const float*)d_in[6];
  const float* Wk = (const float*)d_in[7];
  const float* bk = (const float*)d_in[8];
  const float* Wv = (const float*)d_in[9];
  const float* bv = (const float*)d_in[10];
  const float* Wp = (const float*)d_in[11];
  const float* Wo = (const float*)d_in[12];
  const float* bo = (const float*)d_in[13];
  const float* pbu = (const float*)d_in[14];  // (H,DK) flat == length-D bias
  const float* pbv = (const float*)d_in[15];

  char* ws = (char*)d_ws;
  const size_t NE = (size_t)B_ * T_ * D_;  // 4,194,304 elems
  __bf16* quw = (__bf16*)(ws);
  __bf16* qvw = (__bf16*)(ws + 2 * NE);
  __bf16* kw  = (__bf16*)(ws + 4 * NE);
  __bf16* vw  = (__bf16*)(ws + 6 * NE);
  __bf16* pw  = (__bf16*)(ws + 8 * NE);
  __bf16* xw  = (__bf16*)(ws + 8 * NE + 2 * (size_t)T_ * D_);
  // total ws use: ~43 MB

  dim3 blk(256);
  dim3 g_big(D_ / 64, (B_ * T_) / 128);  // (8, 64)
  dim3 g_pos(D_ / 64, T_ / 128);         // (8, 8)

  // q_u = query@Wq^T + bq + pos_bias_u ; q_v = ... + pos_bias_v
  proj_gemm_k<0, 3><<<g_big, blk, 0, stream>>>(query, nullptr, Wq, bq, pbu,
                                               pbv, quw, qvw, nullptr);
  proj_gemm_k<0, 1><<<g_big, blk, 0, stream>>>(key_, nullptr, Wk, bk, nullptr,
                                               nullptr, kw, nullptr, nullptr);
  proj_gemm_k<0, 1><<<g_big, blk, 0, stream>>>(value, nullptr, Wv, bv, nullptr,
                                               nullptr, vw, nullptr, nullptr);
  proj_gemm_k<0, 1><<<g_pos, blk, 0, stream>>>(pos, nullptr, Wp, nullptr,
                                               nullptr, nullptr, pw, nullptr,
                                               nullptr);

  attn_k<<<dim3(T_ / 64, B_ * H_), dim3(128), 0, stream>>>(quw, qvw, kw, pw,
                                                           vw, maskp, xw);

  // out = x @ Wo^T + bo  (f32 output)
  proj_gemm_k<1, 4><<<g_big, blk, 0, stream>>>(nullptr, xw, Wo, bo, nullptr,
                                               nullptr, nullptr, nullptr,
                                               (float*)d_out);
}
